// RadianceTransformer2_2388001817340
// MI455X (gfx1250) — matrix operand order repaired and checked by reading, not compiled, not run on hardware
//
#include <hip/hip_runtime.h>

typedef __attribute__((ext_vector_type(16))) _Float16 v16h;
typedef __attribute__((ext_vector_type(8)))  _Float16 v8h;
typedef __attribute__((ext_vector_type(8)))  float    v8f;

// ---------------- ws (scratch) layout: f16 transposed weights, [N][K] ----------------
constexpr int OFF_W1T = 0;                     // [64][512]
constexpr int OFF_WQT = 32768;                 // 4 x [256][64]
constexpr int OFF_WKT = OFF_WQT + 4 * 16384;
constexpr int OFF_WVT = OFF_WKT + 4 * 16384;
constexpr int OFF_WOT = OFF_WVT + 4 * 16384;   // 4 x [64][256]
constexpr int OFF_F1T = OFF_WOT + 4 * 16384;   // 4 x [64][64]
constexpr int OFF_F2T = OFF_F1T + 4 * 4096;
constexpr int OFF_CQT = OFF_F2T + 4 * 4096;    // [256][64]
constexpr int OFF_CKT = OFF_CQT + 16384;
constexpr int OFF_CVT = OFF_CKT + 16384;
constexpr int OFF_COT = OFF_CVT + 16384;       // [64][256]

// ---------------- per-wave LDS layout (halfs) ----------------
constexpr int LDSW   = 9728;   // 19456 bytes per wave
constexpr int QS_OFF = 1024;   // X: [16][64]
constexpr int KS_OFF = 2048;   // QS/KS: [16][64]
constexpr int VT_OFF = 3072;   // VT: [64][32] (cols 16..31 zero)
constexpr int PS_OFF = 5120;   // PS: [16][32] (cols 16..31 zero)
constexpr int AO_OFF = 5632;   // AO: [16][256]

__device__ __forceinline__ v8f wmma_f16(v16h a, v16h b, v8f c) {
  return __builtin_amdgcn_wmma_f32_16x16x32_f16(false, a, false, b, (short)0, c,
                                                false, false);
}
__device__ __forceinline__ v8f zero8f() {
  v8f r;
#pragma unroll
  for (int i = 0; i < 8; i++) r[i] = 0.f;
  return r;
}
__device__ __forceinline__ v16h pack16(v8h lo, v8h hi) {
  v16h r;
#pragma unroll
  for (int i = 0; i < 8; i++) { r[i] = lo[i]; r[i + 8] = hi[i]; }
  return r;
}
// A fragment (16x32 f16) from row-major [16][ldk] halfs, k-window at kofs
__device__ __forceinline__ v16h ldA(const _Float16* a, int ldk, int kofs, int lane) {
  int m = lane & 15, kh = kofs + ((lane >> 4) << 3);
  const _Float16* p = a + m * ldk + kh;
  return pack16(*(const v8h*)p, *(const v8h*)(p + 16));
}
// B fragment (32x16 f16) from B^T row-major [N][ldk] halfs
__device__ __forceinline__ v16h ldBT(const _Float16* bt, int ldk, int nbase, int kofs,
                                     int lane) {
  int n = nbase + (lane & 15), k = kofs + ((lane >> 4) << 3);
  const _Float16* p = bt + (long)n * ldk + k;
  return pack16(*(const v8h*)p, *(const v8h*)(p + 16));
}
__device__ __forceinline__ float redsum16(float x) {
#pragma unroll
  for (int m = 1; m < 16; m <<= 1) x += __shfl_xor(x, m, 32);
  return x;
}
__device__ __forceinline__ float redmax16(float x) {
#pragma unroll
  for (int m = 1; m < 16; m <<= 1) x = fmaxf(x, __shfl_xor(x, m, 32));
  return x;
}
// C-fragment -> row-major f16 staging [16][ld] at column base cb
__device__ __forceinline__ void stRow(_Float16* dst, int ld, int cb, v8f acc,
                                      float bias, int lane, bool relu) {
  int hb = lane >> 4, ln = lane & 15;
#pragma unroll
  for (int v = 0; v < 8; v++) {
    float x = acc[v] + bias;
    if (relu) x = fmaxf(x, 0.f);
    dst[(v + 8 * hb) * ld + cb + ln] = (_Float16)x;
  }
}
// C-fragment of V projection -> transposed VT[64][32] (rows = d-local, cols = (b,k))
__device__ __forceinline__ void stVT(_Float16* VT, int cb, v8f acc, float bias,
                                     int lane) {
  int hb = lane >> 4, ln = lane & 15;
  v8h t;
#pragma unroll
  for (int v = 0; v < 8; v++) t[v] = (_Float16)(acc[v] + bias);
  *(v8h*)(VT + (cb + ln) * 32 + 8 * hb) = t;
}
// A fragment straight from f32 global rows (latent), per-lane row pointer
__device__ __forceinline__ v16h gAf32(const float* row, int kofs, int lane) {
  int kh = kofs + ((lane >> 4) << 3);
  const float4* p0 = (const float4*)(row + kh);
  const float4* p1 = (const float4*)(row + kh + 16);
  float4 a0 = p0[0], a1 = p0[1], b0 = p1[0], b1 = p1[1];
  v16h r;
  r[0] = (_Float16)a0.x;  r[1] = (_Float16)a0.y;  r[2] = (_Float16)a0.z;  r[3] = (_Float16)a0.w;
  r[4] = (_Float16)a1.x;  r[5] = (_Float16)a1.y;  r[6] = (_Float16)a1.z;  r[7] = (_Float16)a1.w;
  r[8] = (_Float16)b0.x;  r[9] = (_Float16)b0.y;  r[10] = (_Float16)b0.z; r[11] = (_Float16)b0.w;
  r[12] = (_Float16)b1.x; r[13] = (_Float16)b1.y; r[14] = (_Float16)b1.z; r[15] = (_Float16)b1.w;
  return r;
}
__device__ __forceinline__ void layer_norm(const float (&val)[4][8], const float* g,
                                           const float* be, int ln,
                                           float (&out)[4][8]) {
  float gj[4], bj[4];
#pragma unroll
  for (int j = 0; j < 4; j++) { gj[j] = g[j * 16 + ln]; bj[j] = be[j * 16 + ln]; }
#pragma unroll
  for (int v = 0; v < 8; v++) {
    float s = val[0][v] + val[1][v] + val[2][v] + val[3][v];
    s = redsum16(s);
    float mu = s * (1.f / 64.f);
    float q = 0.f;
#pragma unroll
    for (int j = 0; j < 4; j++) { float d = val[j][v] - mu; q += d * d; }
    q = redsum16(q);
    float inv = rsqrtf(q * (1.f / 64.f) + 1e-5f);
#pragma unroll
    for (int j = 0; j < 4; j++) out[j][v] = (val[j][v] - mu) * inv * gj[j] + bj[j];
  }
}

// ---------------- prep: W[K][N] f32 -> W^T[N][K] f16 in ws ----------------
__global__ void wT_kernel(const float* __restrict__ src, _Float16* __restrict__ dst,
                          int K, int N) {
  int idx = blockIdx.x * blockDim.x + threadIdx.x;
  if (idx >= K * N) return;
  int k = idx / N, n = idx - k * N;
  dst[n * K + k] = (_Float16)src[idx];
}

struct RTParams {
  const float *query, *latent;
  const float *b1, *bq, *bk, *bv, *bo, *fb1, *fb2, *g1, *be1, *g2, *be2;
  const float *cbq, *cbk, *cbv, *cbo, *Wc, *bc, *Ws, *bs;
  const _Float16* ws;
  float *colorO, *sigmaO, *outO, *attnO;
};

__global__ __launch_bounds__(64) void rt_main(RTParams P) {
  const int lane = threadIdx.x & 31;
  const int wid = threadIdx.x >> 5;
  const int hb = lane >> 4;       // which batch of the pair this half-wave maps to
  const int ln = lane & 15;       // column lane within C fragments
  const long b0 = ((long)blockIdx.x * 2 + wid) * 2;
  const bool diag = (hb == ((lane >> 3) & 1));  // same-batch (q,k) lanes

  __shared__ __align__(16) _Float16 smem[2 * LDSW];
  _Float16* X  = smem + wid * LDSW;
  _Float16* QS = X + QS_OFF;
  _Float16* KS = X + KS_OFF;
  _Float16* VT = X + VT_OFF;
  _Float16* PS = X + PS_OFF;
  _Float16* AO = X + AO_OFF;

  // zero pads (K=16..31 of VT and PS stay zero forever)
  {
    v8h z;
#pragma unroll
    for (int i = 0; i < 8; i++) z[i] = (_Float16)0;
#pragma unroll
    for (int r = 0; r < 2; r++) {
      int rr = lane + r * 32;
      *(v8h*)(VT + rr * 32 + 16) = z;
      *(v8h*)(VT + rr * 32 + 24) = z;
    }
    if (lane < 16) { *(v8h*)(PS + ln * 32 + 16) = z; *(v8h*)(PS + ln * 32 + 24) = z; }
  }

  float xr[4][8];  // running activation [16][64] in C-fragment layout

  // ---------------- embed: x = relu(latent @ W1 + b1) ----------------
  {
    const float* lrow = P.latent + ((b0 + (ln >> 3)) * 8 + (ln & 7)) * 512;
    v8f acc[4];
#pragma unroll
    for (int j = 0; j < 4; j++) acc[j] = zero8f();
    for (int t = 0; t < 16; t++) {
      v16h a = gAf32(lrow, t * 32, lane);
#pragma unroll
      for (int j = 0; j < 4; j++)
        acc[j] = wmma_f16(a, ldBT(P.ws + OFF_W1T, 512, j * 16, t * 32, lane), acc[j]);
    }
#pragma unroll
    for (int j = 0; j < 4; j++) {
      float bias = P.b1[j * 16 + ln];
#pragma unroll
      for (int v = 0; v < 8; v++) xr[j][v] = fmaxf(acc[j][v] + bias, 0.f);
    }
  }
  auto storeX = [&]() {
#pragma unroll
    for (int j = 0; j < 4; j++)
#pragma unroll
      for (int v = 0; v < 8; v++)
        X[(v + 8 * hb) * 64 + j * 16 + ln] = (_Float16)xr[j][v];
  };
  storeX();

  // ---------------- transformer layers ----------------
  for (int li = 0; li < 4; li++) {
    const _Float16* WQT = P.ws + OFF_WQT + (long)li * 16384;
    const _Float16* WKT = P.ws + OFF_WKT + (long)li * 16384;
    const _Float16* WVT = P.ws + OFF_WVT + (long)li * 16384;
    const _Float16* WOT = P.ws + OFF_WOT + (long)li * 16384;
    const _Float16* F1T = P.ws + OFF_F1T + (long)li * 4096;
    const _Float16* F2T = P.ws + OFF_F2T + (long)li * 4096;

    v16h aX0 = ldA(X, 64, 0, lane), aX1 = ldA(X, 64, 32, lane);

    for (int h = 0; h < 4; h++) {
      // per-head Q/K/V projections (N = head's 64-col slice)
#pragma unroll
      for (int j2 = 0; j2 < 4; j2++) {
        int nb = h * 64 + j2 * 16;
        v8f q = wmma_f16(aX0, ldBT(WQT, 64, nb, 0, lane), zero8f());
        q = wmma_f16(aX1, ldBT(WQT, 64, nb, 32, lane), q);
        stRow(QS, 64, j2 * 16, q, P.bq[li * 256 + nb + ln], lane, false);
        v8f k = wmma_f16(aX0, ldBT(WKT, 64, nb, 0, lane), zero8f());
        k = wmma_f16(aX1, ldBT(WKT, 64, nb, 32, lane), k);
        stRow(KS, 64, j2 * 16, k, P.bk[li * 256 + nb + ln], lane, false);
        v8f w = wmma_f16(aX0, ldBT(WVT, 64, nb, 0, lane), zero8f());
        w = wmma_f16(aX1, ldBT(WVT, 64, nb, 32, lane), w);
        stVT(VT, j2 * 16, w, P.bv[li * 256 + nb + ln], lane);
      }
      // scores = Q K^T (K rows double as B^T)
      v8f sc = wmma_f16(ldA(QS, 64, 0, lane), ldBT(KS, 64, 0, 0, lane), zero8f());
      sc = wmma_f16(ldA(QS, 64, 32, lane), ldBT(KS, 64, 0, 32, lane), sc);
      // masked softmax (off-diagonal 8x8 batch blocks -> 0)
      float p8[8];
#pragma unroll
      for (int v = 0; v < 8; v++) p8[v] = diag ? sc[v] * 0.125f : -3.0e38f;
#pragma unroll
      for (int v = 0; v < 8; v++) {
        float mx = redmax16(p8[v]);
        float e = diag ? __expf(p8[v] - mx) : 0.f;
        float s = redsum16(e);
        p8[v] = e / s;
      }
      if (diag) {
#pragma unroll
        for (int v = 0; v < 8; v++)
          P.attnO[((((b0 + hb) * 4 + li) * 4 + h) * 8 + v) * 8 + (lane & 7)] = p8[v];
      }
#pragma unroll
      for (int v = 0; v < 8; v++)
        PS[(v + 8 * hb) * 32 + ln] = (_Float16)p8[v];
      // P @ V (K padded to 32 with zeros)
      v16h aP = ldA(PS, 32, 0, lane);
#pragma unroll
      for (int j2 = 0; j2 < 4; j2++) {
        v8f o = wmma_f16(aP, ldBT(VT, 32, j2 * 16, 0, lane), zero8f());
        stRow(AO, 256, h * 64 + j2 * 16, o, 0.f, lane, false);
      }
    }
    // Wo + residual + LN1
    v16h aA[8];
    for (int t = 0; t < 8; t++) aA[t] = ldA(AO, 256, t * 32, lane);
    float val[4][8];
#pragma unroll
    for (int j = 0; j < 4; j++) {
      v8f acc = zero8f();
      for (int t = 0; t < 8; t++)
        acc = wmma_f16(aA[t], ldBT(WOT, 256, j * 16, t * 32, lane), acc);
      float bias = P.bo[li * 64 + j * 16 + ln];
#pragma unroll
      for (int v = 0; v < 8; v++) val[j][v] = acc[v] + bias + xr[j][v];
    }
    layer_norm(val, &P.g1[li * 64], &P.be1[li * 64], ln, xr);
    storeX();
    // FFN: relu(x fW1 + fb1) fW2 + fb2, residual, LN2
    v16h f0 = ldA(X, 64, 0, lane), f1 = ldA(X, 64, 32, lane);
#pragma unroll
    for (int j = 0; j < 4; j++) {
      v8f acc = wmma_f16(f0, ldBT(F1T, 64, j * 16, 0, lane), zero8f());
      acc = wmma_f16(f1, ldBT(F1T, 64, j * 16, 32, lane), acc);
      stRow(QS, 64, j * 16, acc, P.fb1[li * 64 + j * 16 + ln], lane, true);
    }
    v16h g0 = ldA(QS, 64, 0, lane), g1h = ldA(QS, 64, 32, lane);
#pragma unroll
    for (int j = 0; j < 4; j++) {
      v8f acc = wmma_f16(g0, ldBT(F2T, 64, j * 16, 0, lane), zero8f());
      acc = wmma_f16(g1h, ldBT(F2T, 64, j * 16, 32, lane), acc);
      float bias = P.fb2[li * 64 + j * 16 + ln];
#pragma unroll
      for (int v = 0; v < 8; v++) val[j][v] = acc[v] + bias + xr[j][v];
    }
    layer_norm(val, &P.g2[li * 64], &P.be2[li * 64], ln, xr);
    storeX();
  }

  // ---------------- out + sigma ----------------
#pragma unroll
  for (int j = 0; j < 4; j++)
#pragma unroll
    for (int v = 0; v < 8; v++)
      P.outO[((b0 + hb) * 8 + v) * 64 + j * 16 + ln] = xr[j][v];
  {
    float s = 0.f;
#pragma unroll
    for (int j = 0; j < 4; j++) {
      float m = xr[j][0];
#pragma unroll
      for (int v = 1; v < 8; v++) m = fmaxf(m, xr[j][v]);
      s += m * P.Ws[j * 16 + ln];
    }
    s = redsum16(s);
    if (ln == 0) P.sigmaO[b0 + hb] = s + P.bs[0];
  }

  // ---------------- cross attention + color ----------------
  {
    v16h aX0 = ldA(X, 64, 0, lane), aX1 = ldA(X, 64, 32, lane);
    v16h aQ0, aQ1;  // query rows in M positions 0 and 8, all other rows zero
    {
      const float* qp = P.query + (b0 + (ln >> 3)) * 64;
      bool vr = ((ln & 7) == 0);
      int kh = (lane >> 4) << 3;
#pragma unroll
      for (int i = 0; i < 8; i++) {
        aQ0[i]     = vr ? (_Float16)qp[kh + i]      : (_Float16)0;
        aQ0[i + 8] = vr ? (_Float16)qp[kh + 16 + i] : (_Float16)0;
        aQ1[i]     = vr ? (_Float16)qp[32 + kh + i]      : (_Float16)0;
        aQ1[i + 8] = vr ? (_Float16)qp[32 + kh + 16 + i] : (_Float16)0;
      }
    }
    const _Float16* CQT = P.ws + OFF_CQT;
    const _Float16* CKT = P.ws + OFF_CKT;
    const _Float16* CVT = P.ws + OFF_CVT;
    const _Float16* COT = P.ws + OFF_COT;
    for (int h = 0; h < 4; h++) {
#pragma unroll
      for (int j2 = 0; j2 < 4; j2++) {
        int nb = h * 64 + j2 * 16;
        v8f q = wmma_f16(aQ0, ldBT(CQT, 64, nb, 0, lane), zero8f());
        q = wmma_f16(aQ1, ldBT(CQT, 64, nb, 32, lane), q);
        stRow(QS, 64, j2 * 16, q, P.cbq[nb + ln], lane, false);
        v8f k = wmma_f16(aX0, ldBT(CKT, 64, nb, 0, lane), zero8f());
        k = wmma_f16(aX1, ldBT(CKT, 64, nb, 32, lane), k);
        stRow(KS, 64, j2 * 16, k, P.cbk[nb + ln], lane, false);
        v8f w = wmma_f16(aX0, ldBT(CVT, 64, nb, 0, lane), zero8f());
        w = wmma_f16(aX1, ldBT(CVT, 64, nb, 32, lane), w);
        stVT(VT, j2 * 16, w, P.cbv[nb + ln], lane);
      }
      v8f sc = wmma_f16(ldA(QS, 64, 0, lane), ldBT(KS, 64, 0, 0, lane), zero8f());
      sc = wmma_f16(ldA(QS, 64, 32, lane), ldBT(KS, 64, 0, 32, lane), sc);
      float p8[8];
#pragma unroll
      for (int v = 0; v < 8; v++) p8[v] = diag ? sc[v] * 0.125f : -3.0e38f;
#pragma unroll
      for (int v = 0; v < 8; v++) {
        float mx = redmax16(p8[v]);
        float e = diag ? __expf(p8[v] - mx) : 0.f;
        float s = redsum16(e);
        p8[v] = e / s;
      }
#pragma unroll
      for (int v = 0; v < 8; v++)
        PS[(v + 8 * hb) * 32 + ln] = (_Float16)p8[v];
      v16h aP = ldA(PS, 32, 0, lane);
#pragma unroll
      for (int j2 = 0; j2 < 4; j2++) {
        v8f o = wmma_f16(aP, ldBT(VT, 32, j2 * 16, 0, lane), zero8f());
        stRow(AO, 256, h * 64 + j2 * 16, o, 0.f, lane, false);
      }
    }
    v16h aA[8];
    for (int t = 0; t < 8; t++) aA[t] = ldA(AO, 256, t * 32, lane);
    float cav[4];
#pragma unroll
    for (int j = 0; j < 4; j++) {
      v8f acc = zero8f();
      for (int t = 0; t < 8; t++)
        acc = wmma_f16(aA[t], ldBT(COT, 256, j * 16, t * 32, lane), acc);
      cav[j] = fmaxf(acc[0] + P.cbo[j * 16 + ln], 0.f);  // row m=0 or m=8 -> v=0
    }
#pragma unroll
    for (int c = 0; c < 3; c++) {
      float s = 0.f;
#pragma unroll
      for (int j = 0; j < 4; j++) s += cav[j] * P.Wc[(j * 16 + ln) * 3 + c];
      s = redsum16(s);
      if (ln == 0) P.colorO[(b0 + hb) * 3 + c] = s + P.bc[c];
    }
  }
}

extern "C" void kernel_launch(void* const* d_in, const int* in_sizes, int n_in,
                              void* d_out, int out_size, void* d_ws, size_t ws_size,
                              hipStream_t stream) {
  const long B = in_sizes[0] / 64;  // query is [B,1,64]
  _Float16* ws = (_Float16*)d_ws;

  auto T = [&](const void* src, long off, int K, int N) {
    int n = K * N;
    wT_kernel<<<dim3((n + 255) / 256), dim3(256), 0, stream>>>((const float*)src,
                                                               ws + off, K, N);
  };
  T(d_in[2], OFF_W1T, 512, 64);
  for (int i = 0; i < 4; i++) {
    T((const float*)d_in[4]  + (long)i * 64 * 256, OFF_WQT + (long)i * 16384, 64, 256);
    T((const float*)d_in[6]  + (long)i * 64 * 256, OFF_WKT + (long)i * 16384, 64, 256);
    T((const float*)d_in[8]  + (long)i * 64 * 256, OFF_WVT + (long)i * 16384, 64, 256);
    T((const float*)d_in[10] + (long)i * 256 * 64, OFF_WOT + (long)i * 16384, 256, 64);
    T((const float*)d_in[12] + (long)i * 64 * 64,  OFF_F1T + (long)i * 4096, 64, 64);
    T((const float*)d_in[14] + (long)i * 64 * 64,  OFF_F2T + (long)i * 4096, 64, 64);
  }
  T(d_in[20], OFF_CQT, 64, 256);
  T(d_in[22], OFF_CKT, 64, 256);
  T(d_in[24], OFF_CVT, 64, 256);
  T(d_in[26], OFF_COT, 256, 64);

  float* ob = (float*)d_out;
  RTParams P;
  P.query = (const float*)d_in[0];  P.latent = (const float*)d_in[1];
  P.b1  = (const float*)d_in[3];
  P.bq  = (const float*)d_in[5];   P.bk  = (const float*)d_in[7];
  P.bv  = (const float*)d_in[9];   P.bo  = (const float*)d_in[11];
  P.fb1 = (const float*)d_in[13];  P.fb2 = (const float*)d_in[15];
  P.g1  = (const float*)d_in[16];  P.be1 = (const float*)d_in[17];
  P.g2  = (const float*)d_in[18];  P.be2 = (const float*)d_in[19];
  P.cbq = (const float*)d_in[21];  P.cbk = (const float*)d_in[23];
  P.cbv = (const float*)d_in[25];  P.cbo = (const float*)d_in[27];
  P.Wc  = (const float*)d_in[28];  P.bc  = (const float*)d_in[29];
  P.Ws  = (const float*)d_in[30];  P.bs  = (const float*)d_in[31];
  P.ws = ws;
  P.colorO = ob;
  P.sigmaO = ob + 3 * B;
  P.outO   = ob + 4 * B;
  P.attnO  = ob + 4 * B + 512 * B;

  long nblk = (B + 3) / 4;  // 2 waves/block, 2 batch elems/wave
  rt_main<<<dim3((unsigned)nblk), dim3(64), 0, stream>>>(P);
}
